// GsGLstmL_77068893159876
// MI455X (gfx1250) — matrix operand order repaired and checked by reading, not compiled, not run on hardware
//
#include <hip/hip_runtime.h>
#include <stdint.h>

// ---------------------------------------------------------------------------
// GsGLstm on MI455X (gfx1250): GEMMs via v_wmma_f32_16x16x32_bf16 (wave32),
// B tiles staged through LDS with gfx1250 async-to-LDS copies (ASYNCcnt).
// ---------------------------------------------------------------------------

typedef __attribute__((ext_vector_type(16))) __bf16 v16bf;
typedef __attribute__((ext_vector_type(8)))  float  v8f;
typedef int v4i_g __attribute__((vector_size(16)));   // matches builtin param type

#define Bsz   32
#define Nn    512
#define Kn    8
#define Hh    256
#define Ee    64
#define Dd    300
#define Ll    3
#define ROWS  (Bsz * Nn)     // 16384
#define KPAD  320            // D padded to 320; also H+E = 320
#define GCOLS 1024           // 4 gates * H

static __device__ __forceinline__ unsigned short f2bf(float f) {
  union { float f; unsigned u; } v; v.f = f;
  unsigned r = v.u + 0x7FFFu + ((v.u >> 16) & 1u);   // round-to-nearest-even
  return (unsigned short)(r >> 16);
}

// ---- gfx1250 async global->LDS copy (ASYNCcnt) ----------------------------
#if defined(__has_builtin) && __has_builtin(__builtin_amdgcn_global_load_async_to_lds_b128)
static __device__ __forceinline__ void async_copy_b128(const void* g, void* l) {
  __builtin_amdgcn_global_load_async_to_lds_b128(
      (__attribute__((address_space(1))) v4i_g*)(uintptr_t)g,
      (__attribute__((address_space(3))) v4i_g*)(unsigned)(uintptr_t)l,
      0, 0);
}
#else
static __device__ __forceinline__ void async_copy_b128(const void* g, void* l) {
  unsigned la = (unsigned)(uintptr_t)l;   // low 32 bits of generic LDS addr
  asm volatile("global_load_async_to_lds_b128 %0, %1, off"
               :: "v"(la), "v"(g) : "memory");
}
#endif

static __device__ __forceinline__ void wait_async0(void) {
#if defined(__has_builtin) && __has_builtin(__builtin_amdgcn_s_wait_asynccnt)
  __builtin_amdgcn_s_wait_asynccnt(0);
#else
  asm volatile("s_wait_asynccnt 0x0" ::: "memory");
#endif
}

// ---------------------------------------------------------------------------
// GEMM: C[M x ncols] (f32 or bf16) = A_bf16[M x K] @ Bpacked_bf16[K x ncols] + bias
// Block = 256 threads = 8 waves -> 128x64 output tile; wave -> 16x64.
// B is packed per (32x16) tile (each lane's 16 elements contiguous, 32B).
// Per k-step the block needs 4 tiles = 4KB of B, shared by all 8 waves:
// staged via async-to-LDS (1 x b128 per thread), double-buffered, read back
// with ds_load_b128. A rows are per-wave unique -> direct global_load_b128.
// Exact tiling: M%128==0, ncols%64==0, K%32==0 -> EXEC all ones (WMMA req).
// ---------------------------------------------------------------------------
__global__ __launch_bounds__(256)
void gemm_bf16_wmma(const unsigned short* __restrict__ A, int lda,
                    const unsigned short* __restrict__ Bp, int ncols,
                    const float* __restrict__ bias,
                    float* __restrict__ Cf, unsigned short* __restrict__ Cb,
                    int ldc, int coff, int M, int K)
{
  __shared__ __align__(16) unsigned short sB[2][4 * 512];   // 2 x 4KB

  const int tid  = threadIdx.x;
  const int lane = tid & 31;
  const int wave = tid >> 5;
  const int m0 = blockIdx.y * 128 + wave * 16;
  const int n0 = blockIdx.x * 64;
  const int half = lane >> 4;            // 0: lanes 0-15, 1: lanes 16-31
  const int lm   = lane & 15;
  const int koff = half * 16;
  const int ntiles = ncols >> 4;
  const int nsteps = K >> 5;

  // cooperative B staging: thread -> one 16B chunk (4 tiles x 64 chunks = 256)
  const int jj = tid >> 6;               // n-tile 0..3
  const int cc = tid & 63;               // 16B chunk within tile
  const unsigned short* gB = Bp + (((size_t)((n0 >> 4) + jj)) << 9) + (cc << 3);
  const size_t gBstep = (size_t)ntiles << 9;     // elements per k-step
  const int ldsOff = (jj << 9) + (cc << 3);      // element offset in buffer

  async_copy_b128(gB, &sB[0][ldsOff]);           // prologue: stage k-step 0

  v8f acc[4] = {};
  const unsigned short* Arow = A + (size_t)(m0 + lm) * lda + koff;

  for (int kt = 0; kt < nsteps; ++kt) {
    wait_async0();          // own async copy for this k-step landed
    __syncthreads();        // everyone's copy landed; prev reads of other buf done
    if (kt + 1 < nsteps)
      async_copy_b128(gB + (size_t)(kt + 1) * gBstep, &sB[(kt + 1) & 1][ldsOff]);

    // A fragment (16x32 bf16): lane lm holds row m0+lm.
    //   elems 0..15 -> k0+koff+0..15 (koff = 16*half)
    union { v16bf v; uint4 u[2]; } a;
    const unsigned short* ak = Arow + (kt << 5);
    a.u[0] = *(const uint4*)(ak);
    a.u[1] = *(const uint4*)(ak + 8);

    const unsigned short* sb = &sB[kt & 1][lane << 4];   // lane-contig 32B
    #pragma unroll
    for (int j = 0; j < 4; ++j) {
      union { v16bf v; uint4 u[2]; } b;
      const uint4* lp = (const uint4*)(sb + (j << 9));
      b.u[0] = lp[0];
      b.u[1] = lp[1];
      acc[j] = __builtin_amdgcn_wmma_f32_16x16x32_bf16(
          false, a.v, false, b.v, (short)0, acc[j], false, false);
    }
  }

  // C/D layout: VGPR v, lanes 0-15 -> M = v, lanes 16-31 -> M = v+8; N = lane&15
  #pragma unroll
  for (int j = 0; j < 4; ++j) {
    const int n = n0 + j * 16 + lm;
    const float bv = bias ? bias[n] : 0.f;
    #pragma unroll
    for (int v = 0; v < 8; ++v) {
      const int m = m0 + v + half * 8;
      float val = acc[j][v] + bv;
      if (Cf) Cf[(size_t)m * ldc + n] = val;
      if (Cb) Cb[(size_t)m * ldc + coff + n] = f2bf(val);
    }
  }
}

// ---------------------------------------------------------------------------
// Weight packing: fp32 [K x 256] row-major -> packed bf16 (K padded to KPAD)
// packed idx = kt<<13 | nt<<9 | ln<<4 | e ; k = kt*32+(ln>>4)*16+e ; n = nt*16+(ln&15)
// ---------------------------------------------------------------------------
__global__ void pack_b_simple(const float* __restrict__ src, int Ksrc,
                              unsigned short* __restrict__ dst)
{
  int idx = blockIdx.x * blockDim.x + threadIdx.x;   // < KPAD*256 = 81920
  int e  = idx & 15;
  int ln = (idx >> 4) & 31;
  int nt = (idx >> 9) & 15;
  int kt = idx >> 13;
  int n = nt * 16 + (ln & 15);
  int k = kt * 32 + (ln >> 4) * 16 + e;
  float v = (k < Ksrc) ? src[k * 256 + n] : 0.f;
  dst[idx] = f2bf(v);
}

// Stacked gate weights: Wcat[l][src*256+h][g*256+d] from w_in/u_in/w_out/u_out
// (einsum 'bnh,ghd->bngd'), packed per-layer with ntiles = 64.
__global__ void pack_wcat(const float* __restrict__ w_in, const float* __restrict__ u_in,
                          const float* __restrict__ w_out, const float* __restrict__ u_out,
                          unsigned short* __restrict__ dst)
{
  size_t idx = (size_t)blockIdx.x * blockDim.x + threadIdx.x;  // < 3 * 1M
  int l = (int)(idx >> 20);
  int t = (int)(idx & 0xFFFFF);
  int e  = t & 15;
  int ln = (t >> 4) & 31;
  int nt = (t >> 9) & 63;
  int kt = t >> 15;                         // 0..31
  int n = nt * 16 + (ln & 15);              // col c = g*256 + d
  int k = kt * 32 + (ln >> 4) * 16 + e;     // row r = src*256 + h
  int g = n >> 8, d = n & 255;
  int src = k >> 8, h = k & 255;
  const float* p = (src == 0) ? w_in : (src == 1) ? u_in : (src == 2) ? w_out : u_out;
  float v = p[(((size_t)l * 4 + g) * 256 + h) * 256 + d];
  dst[idx] = f2bf(v);
}

__global__ void lemmas_to_bf16(const float* __restrict__ lemmas,
                               unsigned short* __restrict__ A)
{
  int idx = blockIdx.x * blockDim.x + threadIdx.x;   // < ROWS*KPAD
  int k = idx % KPAD;
  int row = idx / KPAD;
  A[idx] = (k < Dd) ? f2bf(lemmas[(size_t)row * Dd + k]) : (unsigned short)0;
}

// Masked neighbor sum of [edge_emb(64) | word_rep(256)] -> bf16 A [ROWS x 320]
__global__ __launch_bounds__(64)
void neighbor_sum(const float* __restrict__ word_rep,    // [ROWS,256]
                  const float* __restrict__ edge_table,  // [64,64]
                  const int* __restrict__ nodes, const int* __restrict__ labels,
                  const float* __restrict__ mask,
                  unsigned short* __restrict__ Aout)
{
  __shared__ int   s_idx[Kn];
  __shared__ int   s_lab[Kn];
  __shared__ float s_m[Kn];
  int node = blockIdx.x;          // b*N + n
  int b = node >> 9;
  if (threadIdx.x < Kn) {
    s_idx[threadIdx.x] = nodes[node * Kn + threadIdx.x];
    s_lab[threadIdx.x] = labels[node * Kn + threadIdx.x];
    s_m[threadIdx.x]   = mask[node * Kn + threadIdx.x];
  }
  __syncthreads();
  for (int col = threadIdx.x; col < KPAD; col += 64) {
    float acc = 0.f;
    if (col < Ee) {
      #pragma unroll
      for (int k = 0; k < Kn; ++k)
        acc += s_m[k] * edge_table[s_lab[k] * Ee + col];
    } else {
      int c = col - Ee;
      #pragma unroll
      for (int k = 0; k < Kn; ++k)
        acc += s_m[k] * word_rep[(size_t)(b * Nn + s_idx[k]) * Hh + c];
    }
    Aout[(size_t)node * KPAD + col] = f2bf(acc);
  }
}

// Masked neighbor sum of hidden -> bf16 into X[row, coff + 0..255]
__global__ __launch_bounds__(256)
void gather_h(const float* __restrict__ hidden, const int* __restrict__ nodes,
              const float* __restrict__ mask, unsigned short* __restrict__ X, int coff)
{
  __shared__ int   s_idx[Kn];
  __shared__ float s_m[Kn];
  int node = blockIdx.x;
  int b = node >> 9;
  if (threadIdx.x < Kn) {
    s_idx[threadIdx.x] = nodes[node * Kn + threadIdx.x];
    s_m[threadIdx.x]   = mask[node * Kn + threadIdx.x];
  }
  __syncthreads();
  int c = threadIdx.x;
  float acc = 0.f;
  #pragma unroll
  for (int k = 0; k < Kn; ++k)
    acc += s_m[k] * hidden[(size_t)(b * Nn + s_idx[k]) * Hh + c];
  X[(size_t)node * GCOLS + coff + c] = f2bf(acc);
}

// LSTM gate nonlinearities + state update (gate order: i, o, f, u)
__global__ __launch_bounds__(256)
void gates(const float* __restrict__ pre, float* __restrict__ cell,
           float* __restrict__ hidden)
{
  int idx = blockIdx.x * blockDim.x + threadIdx.x;   // < ROWS*256
  int row = idx >> 8, d = idx & 255;
  const float* p = pre + (size_t)row * GCOLS;
  float i = 1.f / (1.f + __expf(-p[d]));
  float o = 1.f / (1.f + __expf(-p[256 + d]));
  float f = 1.f / (1.f + __expf(-p[512 + d]));
  float u = tanhf(p[768 + d]);
  float c = f * cell[idx] + i * u;
  cell[idx] = c;
  hidden[idx] = o * tanhf(c);
}

// Final entity gather-sum -> out [B, ENT, H] fp32
__global__ __launch_bounds__(256)
void entity_out(const float* __restrict__ hidden, const int* __restrict__ idxs,
                const float* __restrict__ mask, float* __restrict__ out)
{
  int be = blockIdx.x;            // b*ENT + e, 0..63
  int b = be >> 1;
  int h = threadIdx.x;
  float acc = 0.f;
  #pragma unroll
  for (int m = 0; m < 8; ++m) {
    float mm = mask[be * 8 + m];
    int nd = idxs[be * 8 + m];
    acc += mm * hidden[(size_t)(b * Nn + nd) * Hh + h];
  }
  out[(size_t)be * Hh + h] = acc;
}

// ---------------------------------------------------------------------------
extern "C" void kernel_launch(void* const* d_in, const int* in_sizes, int n_in,
                              void* d_out, int out_size, void* d_ws, size_t ws_size,
                              hipStream_t stream)
{
  const float* lemmas     = (const float*)d_in[0];
  const int*   in_nodes   = (const int*)d_in[1];
  const int*   in_labels  = (const int*)d_in[2];
  const int*   out_nodes  = (const int*)d_in[3];
  const int*   out_labels = (const int*)d_in[4];
  const int*   ent_idx    = (const int*)d_in[5];
  const float* in_mask    = (const float*)d_in[6];
  const float* out_mask   = (const float*)d_in[7];
  const float* ent_mask   = (const float*)d_in[8];
  const float* W_emb      = (const float*)d_in[9];
  const float* b_emb      = (const float*)d_in[10];
  const float* edge_table = (const float*)d_in[11];
  const float* W_edge_rep = (const float*)d_in[12];
  const float* b_edge_rep = (const float*)d_in[13];
  const float* w_in       = (const float*)d_in[14];
  const float* u_in       = (const float*)d_in[15];
  const float* w_out      = (const float*)d_in[16];
  const float* u_out      = (const float*)d_in[17];
  const float* b_g        = (const float*)d_in[18];

  char* ws = (char*)d_ws;
  size_t off = 0;
  auto take = [&](size_t bytes) -> char* {
    char* p = ws + off;
    off = (off + bytes + 255) & ~(size_t)255;
    return p;
  };

  float* hidden = (float*)take((size_t)ROWS * Hh * 4);           // also word_rep
  float* cellb  = (float*)take((size_t)ROWS * Hh * 4);
  unsigned short* X     = (unsigned short*)take((size_t)ROWS * GCOLS * 2);
  float*          pre   = (float*)take((size_t)ROWS * GCOLS * 4);
  unsigned short* Aemb  = (unsigned short*)take((size_t)ROWS * KPAD * 2);
  unsigned short* WembP = (unsigned short*)take((size_t)KPAD * Hh * 2);
  unsigned short* WedgeP= (unsigned short*)take((size_t)KPAD * Hh * 2);
  unsigned short* WcatP = (unsigned short*)take((size_t)Ll * GCOLS * GCOLS * 2);

  (void)hipMemsetAsync(cellb, 0, (size_t)ROWS * Hh * 4, stream);

  // pack weights to bf16 fragment layout
  pack_b_simple<<<(KPAD * Hh) / 256, 256, 0, stream>>>(W_emb, Dd, WembP);
  pack_b_simple<<<(KPAD * Hh) / 256, 256, 0, stream>>>(W_edge_rep, KPAD, WedgeP);
  pack_wcat<<<(Ll * GCOLS * GCOLS) / 256, 256, 0, stream>>>(w_in, u_in, w_out, u_out, WcatP);
  lemmas_to_bf16<<<(ROWS * KPAD) / 256, 256, 0, stream>>>(lemmas, Aemb);

  // word_rep = lemmas @ W_emb + b_emb  -> hidden (fp32)
  dim3 gEmb(Hh / 64, ROWS / 128);   // (4,128)
  gemm_bf16_wmma<<<gEmb, 256, 0, stream>>>(Aemb, KPAD, WembP, Hh, b_emb,
                                           hidden, nullptr, Hh, 0, ROWS, KPAD);

  // in_rep -> X cols [256..511]
  neighbor_sum<<<ROWS, 64, 0, stream>>>(hidden, edge_table, in_nodes, in_labels,
                                        in_mask, Aemb);
  gemm_bf16_wmma<<<gEmb, 256, 0, stream>>>(Aemb, KPAD, WedgeP, Hh, b_edge_rep,
                                           nullptr, X, GCOLS, 256, ROWS, KPAD);
  // out_rep -> X cols [768..1023]
  neighbor_sum<<<ROWS, 64, 0, stream>>>(hidden, edge_table, out_nodes, out_labels,
                                        out_mask, Aemb);
  gemm_bf16_wmma<<<gEmb, 256, 0, stream>>>(Aemb, KPAD, WedgeP, Hh, b_edge_rep,
                                           nullptr, X, GCOLS, 768, ROWS, KPAD);

  // graph-LSTM layers
  dim3 gLay(GCOLS / 64, ROWS / 128);  // (16,128)
  for (int l = 0; l < Ll; ++l) {
    gather_h<<<ROWS, 256, 0, stream>>>(hidden, in_nodes, in_mask, X, 0);
    gather_h<<<ROWS, 256, 0, stream>>>(hidden, out_nodes, out_mask, X, 512);
    gemm_bf16_wmma<<<gLay, 256, 0, stream>>>(X, GCOLS,
                                             WcatP + (size_t)l * GCOLS * GCOLS,
                                             GCOLS, b_g + l * GCOLS,
                                             pre, nullptr, GCOLS, 0, ROWS, GCOLS);
    gates<<<ROWS, 256, 0, stream>>>(pre, cellb, hidden);
  }

  entity_out<<<Bsz * 2, 256, 0, stream>>>(hidden, ent_idx, ent_mask, (float*)d_out);
}
